// GRUCell_6992206758569
// MI455X (gfx1250) — matrix operand order repaired and checked by reading, not compiled, not run on hardware
//
#include <hip/hip_runtime.h>
#include <hip/hip_bf16.h>

// ---------------------------------------------------------------------------
// MAGCN-GRU cell for MI455X (gfx1250, wave32, WMMA).
// Dominant work: 256x (1024x1024)@(1024x128) GEMMs (~69 GFLOP) -> bf16 WMMA
// with f32 accumulate. Per-node weights e@W_pool are generated on the fly in
// LDS (never materialized: saves 300MB of traffic). Activations + GRU combine
// fused into GEMM epilogues.
// ---------------------------------------------------------------------------

typedef __bf16 bf16;
typedef __bf16 v16bf __attribute__((ext_vector_type(16)));
typedef float  v8f   __attribute__((ext_vector_type(8)));

#define NNODE 1024
#define BATCH 64
#define CIN   64
#define HID   64
#define DEMB  16
#define CH    128   // C + H
#define KI    384   // CHEB_K * (C+H)

// ---- WMMA fragment helpers (wave32 layouts per CDNA5 ISA 7.12.2) ----------

// A (16x32 bf16), row-major LDS source: lane holds row M=lane%16; K mapping:
// K = e + 16*half (matches frag_Bt).
__device__ __forceinline__ v16bf frag_A(const bf16* A, int ldk, int m0, int k0, int lane) {
  int m  = m0 + (lane & 15);
  int h16 = (lane >> 4) << 4;
  v16bf a;
#pragma unroll
  for (int e = 0; e < 16; ++e) {
    int k = k0 + h16 + e;
    a[e] = A[m * ldk + k];
  }
  return a;
}

// B (32x16 bf16) from an LDS tile stored transposed Bt[n][k] (ldk = K-extent):
// lane holds column N=lane%16, K = e + 16*(lane/16) -> 16 contiguous bf16.
__device__ __forceinline__ v16bf frag_Bt(const bf16* Bt, int ldk, int n0, int k0, int lane) {
  int n  = n0 + (lane & 15);
  int kb = k0 + ((lane >> 4) << 4);
  v16bf b;
#pragma unroll
  for (int e = 0; e < 16; ++e) b[e] = Bt[n * ldk + kb + e];
  return b;
}

__device__ __forceinline__ v8f wmma_bf16(v16bf a, v16bf b, v8f c) {
  return __builtin_amdgcn_wmma_f32_16x16x32_bf16(false, a, false, b, (short)0, c,
                                                 false, false);
}

// ---- Stage 1: e = LayerNorm(node_emb + time_emb), both param sets ---------
__global__ void k_embed(const float* __restrict__ ne, const float* __restrict__ te,
                        const float* __restrict__ wg, const float* __restrict__ bg,
                        const float* __restrict__ wu, const float* __restrict__ bu,
                        float* __restrict__ eg, float* __restrict__ eu) {
  int t = blockIdx.x * blockDim.x + threadIdx.x;
  int n = t >> 1, path = t & 1;
  if (n >= NNODE) return;
  float v[DEMB]; float m = 0.f;
#pragma unroll
  for (int d = 0; d < DEMB; ++d) { v[d] = ne[n * DEMB + d] + te[d]; m += v[d]; }
  m *= (1.f / DEMB);
  float var = 0.f;
#pragma unroll
  for (int d = 0; d < DEMB; ++d) { float c = v[d] - m; var += c * c; }
  var *= (1.f / DEMB);
  float inv = rsqrtf(var + 1e-12f);
  const float* w = path ? wu : wg;
  const float* b = path ? bu : bg;
  float* out = path ? eu : eg;
#pragma unroll
  for (int d = 0; d < DEMB; ++d) out[n * DEMB + d] = (v[d] - m) * inv * w[d] + b[d];
}

// ---- Stage 2: pack concat(x,state) -> bf16; seed candidate x-half ---------
__global__ void k_convert(const float* __restrict__ x, const float* __restrict__ st,
                          bf16* __restrict__ xcat, bf16* __restrict__ cand) {
  int i = blockIdx.x * blockDim.x + threadIdx.x;
  if (i >= BATCH * NNODE * CIN) return;
  int c = i & 63, bn = i >> 6;
  bf16 xv = (bf16)x[i];
  xcat[bn * CH + c] = xv;
  cand[bn * CH + c] = xv;
  xcat[bn * CH + 64 + c] = (bf16)st[i];
}

// ---- Stage 3: supports = row-softmax(e e^T), one block per row ------------
__global__ void k_supports(const float* __restrict__ e, bf16* __restrict__ sup) {
  int n = blockIdx.x, tid = threadIdx.x;
  __shared__ float er[DEMB];
  __shared__ float red[256];
  if (tid < DEMB) er[tid] = e[n * DEMB + tid];
  __syncthreads();
  float lg[4];
  float mx = -3.4e38f;
#pragma unroll
  for (int j = 0; j < 4; ++j) {
    int m = tid + j * 256;
    float s = 0.f;
#pragma unroll
    for (int d = 0; d < DEMB; ++d) s += er[d] * e[m * DEMB + d];
    lg[j] = s; mx = fmaxf(mx, s);
  }
  red[tid] = mx; __syncthreads();
  for (int s = 128; s > 0; s >>= 1) {
    if (tid < s) red[tid] = fmaxf(red[tid], red[tid + s]);
    __syncthreads();
  }
  mx = red[0]; __syncthreads();
  float sum = 0.f;
#pragma unroll
  for (int j = 0; j < 4; ++j) { lg[j] = __expf(lg[j] - mx); sum += lg[j]; }
  red[tid] = sum; __syncthreads();
  for (int s = 128; s > 0; s >>= 1) {
    if (tid < s) red[tid] += red[tid + s];
    __syncthreads();
  }
  float inv = 1.f / red[0];
#pragma unroll
  for (int j = 0; j < 4; ++j) sup[n * NNODE + tid + j * 256] = (bf16)(lg[j] * inv);
}

// ---- Stage 4: T2 = 2*S@S - I (WMMA, 64x64 tile per block) -----------------
__global__ void k_cheb(const bf16* __restrict__ sg, bf16* __restrict__ tg,
                       const bf16* __restrict__ su, bf16* __restrict__ tu) {
  const bf16* S = blockIdx.z ? su : sg;
  bf16* T = blockIdx.z ? tu : tg;
  int m0b = blockIdx.x * 64, n0b = blockIdx.y * 64;
  __shared__ bf16 As[64 * 32];
  __shared__ bf16 Bt[64 * 32];
  int tid = threadIdx.x, wave = tid >> 5, lane = tid & 31;
  int mt = wave & 3, ntg = (wave >> 2) * 2;
  v8f acc[2] = {};
  for (int k0 = 0; k0 < NNODE; k0 += 32) {
    __syncthreads();
    for (int i = tid; i < 2048; i += 256) {
      int r = i >> 5, c = i & 31;
      As[i] = S[(m0b + r) * NNODE + k0 + c];
    }
    for (int i = tid; i < 2048; i += 256) {
      int k = i >> 6, nn = i & 63;
      Bt[nn * 32 + k] = S[(k0 + k) * NNODE + n0b + nn];
    }
    __syncthreads();
    v16bf a = frag_A(As, 32, mt * 16, 0, lane);
#pragma unroll
    for (int j = 0; j < 2; ++j) {
      v16bf b = frag_Bt(Bt, 32, (ntg + j) * 16, 0, lane);
      acc[j] = wmma_bf16(a, b, acc[j]);
    }
  }
#pragma unroll
  for (int j = 0; j < 2; ++j) {
    int col = n0b + (ntg + j) * 16 + (lane & 15);
    int mb  = m0b + mt * 16 + ((lane >> 4) << 3);
#pragma unroll
    for (int v = 0; v < 8; ++v) {
      int row = mb + v;
      float t = 2.f * acc[j][v] - (row == col ? 1.f : 0.f);
      T[row * NNODE + col] = (bf16)t;
    }
  }
}

// ---- Stage 5: x_g[k,b] = S_k @ X[b]  (64 nodes x 128 ch tile per block) ---
__global__ void k_xg(const bf16* __restrict__ S1, const bf16* __restrict__ S2,
                     const bf16* __restrict__ X, bf16* __restrict__ O1,
                     bf16* __restrict__ O2) {
  const bf16* S = blockIdx.z ? S2 : S1;
  bf16* O = blockIdx.z ? O2 : O1;
  int n0b = blockIdx.x * 64, b = blockIdx.y;
  const bf16* Xb = X + (size_t)b * NNODE * CH;
  bf16* Ob = O + (size_t)b * NNODE * CH;
  __shared__ bf16 As[64 * 32];
  __shared__ bf16 Bt[CH * 32];
  int tid = threadIdx.x, wave = tid >> 5, lane = tid & 31;
  int mt = wave & 3, ntg = (wave >> 2) * 4;
  v8f acc[4] = {};
  for (int k0 = 0; k0 < NNODE; k0 += 32) {
    __syncthreads();
    for (int i = tid; i < 2048; i += 256) {
      int r = i >> 5, c = i & 31;
      As[i] = S[(n0b + r) * NNODE + k0 + c];
    }
    for (int i = tid; i < 4096; i += 256) {
      int m = i >> 7, c = i & 127;
      Bt[c * 32 + m] = Xb[(k0 + m) * CH + c];
    }
    if (k0 + 32 < NNODE) __builtin_prefetch(&S[n0b * NNODE + k0 + 32], 0, 1);
    __syncthreads();
    v16bf a = frag_A(As, 32, mt * 16, 0, lane);
#pragma unroll
    for (int j = 0; j < 4; ++j) {
      v16bf bb = frag_Bt(Bt, 32, (ntg + j) * 16, 0, lane);
      acc[j] = wmma_bf16(a, bb, acc[j]);
    }
  }
#pragma unroll
  for (int j = 0; j < 4; ++j) {
    int c  = (ntg + j) * 16 + (lane & 15);
    int nb = n0b + mt * 16 + ((lane >> 4) << 3);
#pragma unroll
    for (int v = 0; v < 8; ++v) Ob[(nb + v) * CH + c] = (bf16)acc[j][v];
  }
}

// ---- Stage 6: gate magcn, one block per node. W slab = e@gate_W in LDS ----
__global__ void k_gate(const float* __restrict__ eg, const float* __restrict__ gW,
                       const float* __restrict__ gb, const bf16* __restrict__ xcat,
                       const bf16* __restrict__ xg1, const bf16* __restrict__ xg2,
                       const float* __restrict__ state,
                       bf16* __restrict__ cand, float* __restrict__ r_out) {
  int n = blockIdx.x, tid = threadIdx.x;
  __shared__ bf16 As[64 * KI];     // 48 KB: batch(64) x (3*128) activations
  __shared__ bf16 Wt[128 * 32];    // 8 KB:  W K-slab transposed [o][k]
  __shared__ float bias[128];
  __shared__ float esh[DEMB];
  if (tid < DEMB) esh[tid] = eg[n * DEMB + tid];
  __syncthreads();
  if (tid < 128) {
    float s = 0.f;
#pragma unroll
    for (int d = 0; d < DEMB; ++d) s += esh[d] * gb[d * 128 + tid];
    bias[tid] = s;
  }
  for (int i = tid; i < 64 * KI; i += 256) {
    int bb = i / KI, jj = i % KI;
    int kp = jj >> 7, c = jj & 127;
    const bf16* src = (kp == 0) ? xcat : (kp == 1 ? xg1 : xg2);
    As[i] = src[((size_t)bb * NNODE + n) * CH + c];
  }
  int wave = tid >> 5, lane = tid & 31;
  int mt = wave & 3, ntg = (wave >> 2) * 4;
  v8f acc[4] = {};
  for (int k0 = 0; k0 < KI; k0 += 32) {
    __syncthreads();
    for (int i = tid; i < 4096; i += 256) {     // generate W slab: e . W_pool
      int o = i & 127, kl = i >> 7;
      float s = 0.f;
#pragma unroll
      for (int d = 0; d < DEMB; ++d) s += esh[d] * gW[(d * KI + k0 + kl) * 128 + o];
      Wt[o * 32 + kl] = (bf16)s;
    }
    __syncthreads();
    v16bf a = frag_A(As, KI, mt * 16, k0, lane);
#pragma unroll
    for (int j = 0; j < 4; ++j) {
      v16bf b = frag_Bt(Wt, 32, (ntg + j) * 16, 0, lane);
      acc[j] = wmma_bf16(a, b, acc[j]);
    }
  }
#pragma unroll
  for (int j = 0; j < 4; ++j) {
    int o   = (ntg + j) * 16 + (lane & 15);
    int bb0 = mt * 16 + ((lane >> 4) << 3);
#pragma unroll
    for (int v = 0; v < 8; ++v) {
      int bb = bb0 + v;
      float sg = 1.f / (1.f + __expf(-(acc[j][v] + bias[o])));
      size_t bn = (size_t)bb * NNODE + n;
      if (o < HID) {   // z gate -> candidate second half = z * state
        cand[bn * CH + HID + o] = (bf16)(sg * state[bn * HID + o]);
      } else {         // r gate -> scratch for final combine
        r_out[bn * HID + (o - HID)] = sg;
      }
    }
  }
}

// ---- Stage 7: update magcn + GRU combine, one block per node --------------
__global__ void k_update(const float* __restrict__ eu, const float* __restrict__ uW,
                         const float* __restrict__ ub, const bf16* __restrict__ cand,
                         const bf16* __restrict__ xg1, const bf16* __restrict__ xg2,
                         const float* __restrict__ state, const float* __restrict__ r_in,
                         float* __restrict__ out) {
  int n = blockIdx.x, tid = threadIdx.x;
  __shared__ bf16 As[64 * KI];     // 48 KB
  __shared__ bf16 Wt[64 * 32];     // 4 KB
  __shared__ float bias[64];
  __shared__ float esh[DEMB];
  if (tid < DEMB) esh[tid] = eu[n * DEMB + tid];
  __syncthreads();
  if (tid < 64) {
    float s = 0.f;
#pragma unroll
    for (int d = 0; d < DEMB; ++d) s += esh[d] * ub[d * 64 + tid];
    bias[tid] = s;
  }
  for (int i = tid; i < 64 * KI; i += 256) {
    int bb = i / KI, jj = i % KI;
    int kp = jj >> 7, c = jj & 127;
    const bf16* src = (kp == 0) ? cand : (kp == 1 ? xg1 : xg2);
    As[i] = src[((size_t)bb * NNODE + n) * CH + c];
  }
  int wave = tid >> 5, lane = tid & 31;
  int mt = wave & 3, otg = (wave >> 2) * 2;
  v8f acc[2] = {};
  for (int k0 = 0; k0 < KI; k0 += 32) {
    __syncthreads();
    for (int i = tid; i < 2048; i += 256) {
      int o = i & 63, kl = i >> 6;
      float s = 0.f;
#pragma unroll
      for (int d = 0; d < DEMB; ++d) s += esh[d] * uW[(d * KI + k0 + kl) * 64 + o];
      Wt[o * 32 + kl] = (bf16)s;
    }
    __syncthreads();
    v16bf a = frag_A(As, KI, mt * 16, k0, lane);
#pragma unroll
    for (int j = 0; j < 2; ++j) {
      v16bf b = frag_Bt(Wt, 32, (otg + j) * 16, 0, lane);
      acc[j] = wmma_bf16(a, b, acc[j]);
    }
  }
#pragma unroll
  for (int j = 0; j < 2; ++j) {
    int o   = (otg + j) * 16 + (lane & 15);
    int bb0 = mt * 16 + ((lane >> 4) << 3);
#pragma unroll
    for (int v = 0; v < 8; ++v) {
      int bb = bb0 + v;
      float hc = tanhf(acc[j][v] + bias[o]);
      size_t idx = ((size_t)bb * NNODE + n) * HID + o;
      float r = r_in[idx];
      out[idx] = r * state[idx] + (1.f - r) * hc;
    }
  }
}

// ---------------------------------------------------------------------------
extern "C" void kernel_launch(void* const* d_in, const int* in_sizes, int n_in,
                              void* d_out, int out_size, void* d_ws, size_t ws_size,
                              hipStream_t stream) {
  const float* x   = (const float*)d_in[0];
  const float* st  = (const float*)d_in[2];
  const float* ne  = (const float*)d_in[3];
  const float* te  = (const float*)d_in[5];
  const float* gW  = (const float*)d_in[6];
  const float* gb  = (const float*)d_in[7];
  const float* glw = (const float*)d_in[8];
  const float* glb = (const float*)d_in[9];
  const float* uW  = (const float*)d_in[10];
  const float* ub  = (const float*)d_in[11];
  const float* ulw = (const float*)d_in[12];
  const float* ulb = (const float*)d_in[13];
  float* out = (float*)d_out;

  // Workspace carve-up (~92 MB).
  float* eg    = (float*)d_ws;
  float* eu    = eg + NNODE * DEMB;
  bf16* sup_g  = (bf16*)(eu + NNODE * DEMB);
  bf16* t2_g   = sup_g + (size_t)NNODE * NNODE;
  bf16* sup_u  = t2_g + (size_t)NNODE * NNODE;
  bf16* t2_u   = sup_u + (size_t)NNODE * NNODE;
  bf16* xcat   = t2_u + (size_t)NNODE * NNODE;
  bf16* cand   = xcat + (size_t)BATCH * NNODE * CH;
  bf16* xg1    = cand + (size_t)BATCH * NNODE * CH;    // reused for both magcns
  bf16* xg2    = xg1 + (size_t)BATCH * NNODE * CH;
  float* r_buf = (float*)(xg2 + (size_t)BATCH * NNODE * CH);

  k_embed<<<8, 256, 0, stream>>>(ne, te, glw, glb, ulw, ulb, eg, eu);
  k_convert<<<(BATCH * NNODE * CIN + 255) / 256, 256, 0, stream>>>(x, st, xcat, cand);
  k_supports<<<NNODE, 256, 0, stream>>>(eg, sup_g);
  k_supports<<<NNODE, 256, 0, stream>>>(eu, sup_u);
  k_cheb<<<dim3(16, 16, 2), 256, 0, stream>>>(sup_g, t2_g, sup_u, t2_u);

  // gate path
  k_xg<<<dim3(16, 64, 2), 256, 0, stream>>>(sup_g, t2_g, xcat, xg1, xg2);
  k_gate<<<NNODE, 256, 0, stream>>>(eg, gW, gb, xcat, xg1, xg2, st, cand, r_buf);

  // update path (xg buffers reused; stream order guarantees safety)
  k_xg<<<dim3(16, 64, 2), 256, 0, stream>>>(sup_u, t2_u, cand, xg1, xg2);
  k_update<<<NNODE, 256, 0, stream>>>(eu, uW, ub, cand, xg1, xg2, st, r_buf, out);

  (void)in_sizes; (void)n_in; (void)out_size; (void)ws_size;
}
